// CharEncoder_35974646072095
// MI455X (gfx1250) — compile-verified, matrix operand order checked by reading, not run on hardware
//
#include <hip/hip_runtime.h>
#include <hip/hip_bf16.h>
#include <math.h>

// ---------------------------------------------------------------------------
// CharEncoder for MI455X (gfx1250): bf16 WMMA GEMMs + sequential LSTM scan.
//   B=64, S=256, E=H=512, input feature K=400 (padded to 416 = 13*32).
// Register-blocked 32x64 wave tiles for the big GEMMs (launch_bounds relaxes
// the VGPR budget so accumulators never spill); async-to-LDS staged Whh for
// the recurrent step kernel.
// ---------------------------------------------------------------------------

typedef __attribute__((ext_vector_type(16))) __bf16 bf16x16;
typedef __attribute__((ext_vector_type(8)))  float  v8f;

#define B_    64
#define S_    256
#define E_    512
#define H_    512
#define K_IN  416      // (DC+DB)*2 = 400, zero-padded to multiple of 32
#define G4    2048     // 4*H

// ---- WMMA helpers ----------------------------------------------------------

__device__ __forceinline__ v8f wmma_bf16(bf16x16 a, bf16x16 b, v8f c) {
  // D = A(16x32 bf16) * B(32x16 bf16) + C(16x16 f32)
  return __builtin_amdgcn_wmma_f32_16x16x32_bf16(
      /*neg_a=*/false, a, /*neg_b=*/false, b,
      /*c_mod=*/(short)0, c, /*reuse_a=*/false, /*reuse_b=*/false);
}

// Load a 16-row x 32-K bf16 tile in the documented wave32 VGPR striping:
//   lanes 0-15 : elems 0-7 -> K=k0+0..7,  elems 8-15 -> K=k0+16..23
//   lanes 16-31: elems 0-7 -> K=k0+8..15, elems 8-15 -> K=k0+24..31
// Works for A (row-major [M,K]) and W-as-B^T (row-major [N,K]): NT GEMM,
// both operands K-contiguous -> two global_load_b128 per lane.
__device__ __forceinline__ bf16x16 load_tileK(const __bf16* base, int ld,
                                              int row0, int k0) {
  const int lane = threadIdx.x & 31;
  const int r    = row0 + (lane & 15);
  const int half = lane >> 4;
  const __bf16* p = base + (size_t)r * ld + k0 + (half << 3);
  bf16x16 v;
#pragma unroll
  for (int e = 0; e < 8; ++e) v[e] = p[e];
#pragma unroll
  for (int e = 0; e < 8; ++e) v[8 + e] = p[16 + e];
  return v;
}

__device__ __forceinline__ float sigmoidf_(float x) {
  return 1.0f / (1.0f + __expf(-x));
}

// ---- Prep kernels ----------------------------------------------------------

__global__ void k_cvt(const float* __restrict__ src, __bf16* __restrict__ dst,
                      int n) {
  int i = blockIdx.x * blockDim.x + threadIdx.x;
  if (i < n) dst[i] = (__bf16)src[i];
}

__global__ void k_pad_cvt(const float* __restrict__ src,
                          __bf16* __restrict__ dst,
                          int rows, int scols, int dcols) {
  int i = blockIdx.x * blockDim.x + threadIdx.x;
  if (i >= rows * dcols) return;
  int r = i / dcols, c = i - r * dcols;
  dst[i] = (c < scols) ? (__bf16)src[(size_t)r * scols + c] : (__bf16)0.0f;
}

__global__ void k_bsum(const float* __restrict__ a, const float* __restrict__ b,
                       float* __restrict__ o, int n) {
  int i = blockIdx.x * blockDim.x + threadIdx.x;
  if (i < n) o[i] = a[i] + b[i];
}

__global__ void k_zero_bf(__bf16* p, int n) {
  int i = blockIdx.x * blockDim.x + threadIdx.x;
  if (i < n) p[i] = (__bf16)0.0f;
}

__global__ void k_zero_f(float* p, int n) {
  int i = blockIdx.x * blockDim.x + threadIdx.x;
  if (i < n) p[i] = 0.0f;
}

// ---- Embedding gather: xin[s*B+b, 0:416] (bf16, zero-padded) ---------------
// Faithful to reference: bichar uses insts_bichar_l for BOTH halves.

__global__ void k_embed(const int* __restrict__ ic, const int* __restrict__ ibl,
                        const float* __restrict__ cts, const float* __restrict__ ct,
                        const float* __restrict__ bts, const float* __restrict__ bt,
                        __bf16* __restrict__ xin) {
  const int row = blockIdx.x;      // row = s*B + b  (time-major)
  const int s = row >> 6;
  const int b = row & 63;
  const int ci = ic [b * S_ + s];
  const int bi = ibl[b * S_ + s];
  __bf16* dst = xin + (size_t)row * K_IN;
  for (int col = threadIdx.x; col < K_IN; col += blockDim.x) {
    float v;
    if      (col < 100) v = cts[(size_t)ci * 100 + col];
    else if (col < 200) v = ct [(size_t)ci * 100 + (col - 100)];
    else if (col < 300) v = bts[(size_t)bi * 100 + (col - 200)];
    else if (col < 400) v = bt [(size_t)bi * 100 + (col - 300)];
    else                v = 0.0f;
    dst[col] = (__bf16)v;
  }
}

// ---- Projection GEMM: x = tanh(xin @ W^T + b), both dirs -------------------
// M=16384, N=512, K=416. Wave tile 32x64 (2Mx4N accumulators): per k-step
// 6 operand tiles feed 8 WMMAs. launch_bounds(256,1): no accumulator spills.

__global__ void __launch_bounds__(256, 1)
k_proj(const __bf16* __restrict__ xin,
       const __bf16* __restrict__ wl, const __bf16* __restrict__ wr,
       const float* __restrict__ bl,  const float* __restrict__ br,
       __bf16* __restrict__ xl, __bf16* __restrict__ xr) {
  const int wave = blockIdx.x * (blockDim.x >> 5) + (threadIdx.x >> 5);
  const int dir = wave >> 12;          // 4096 wave-tiles per direction
  const int rem = wave & 4095;
  const int m0 = (rem >> 3) << 5;      // 512 M-supertiles of 32
  const int n0 = (rem & 7) << 6;       // 8 N-supertiles of 64
  const __bf16* W    = dir ? wr : wl;
  const float*  bias = dir ? br : bl;
  __bf16*       X    = dir ? xr : xl;

  v8f acc[2][4];
#pragma unroll
  for (int mi = 0; mi < 2; ++mi)
#pragma unroll
    for (int ni = 0; ni < 4; ++ni)
      acc[mi][ni] = v8f{0.f, 0.f, 0.f, 0.f, 0.f, 0.f, 0.f, 0.f};

  for (int k0 = 0; k0 < K_IN; k0 += 32) {
    bf16x16 a0 = load_tileK(xin, K_IN, m0, k0);
    bf16x16 a1 = load_tileK(xin, K_IN, m0 + 16, k0);
    bf16x16 b0 = load_tileK(W, K_IN, n0 +  0, k0);
    bf16x16 b1 = load_tileK(W, K_IN, n0 + 16, k0);
    bf16x16 b2 = load_tileK(W, K_IN, n0 + 32, k0);
    bf16x16 b3 = load_tileK(W, K_IN, n0 + 48, k0);
    acc[0][0] = wmma_bf16(a0, b0, acc[0][0]);
    acc[0][1] = wmma_bf16(a0, b1, acc[0][1]);
    acc[0][2] = wmma_bf16(a0, b2, acc[0][2]);
    acc[0][3] = wmma_bf16(a0, b3, acc[0][3]);
    acc[1][0] = wmma_bf16(a1, b0, acc[1][0]);
    acc[1][1] = wmma_bf16(a1, b1, acc[1][1]);
    acc[1][2] = wmma_bf16(a1, b2, acc[1][2]);
    acc[1][3] = wmma_bf16(a1, b3, acc[1][3]);
  }

  const int lane = threadIdx.x & 31;
  const int half = lane >> 4;
#pragma unroll
  for (int ni = 0; ni < 4; ++ni) {
    const int col = n0 + ni * 16 + (lane & 15);
    const float bv = bias[col];
#pragma unroll
    for (int mi = 0; mi < 2; ++mi) {
#pragma unroll
      for (int r = 0; r < 8; ++r) {
        int row = m0 + mi * 16 + r + (half << 3);
        X[(size_t)row * E_ + col] = (__bf16)tanhf(acc[mi][ni][r] + bv);
      }
    }
  }
}

// ---- Gate-preactivation GEMM: gi = x @ Wih^T + (bih + bhh) -----------------
// M=16384, N=2048, K=512, fp32 out. Wave tile 32x64. 32768 waves -> 4096 blk.

__global__ void __launch_bounds__(256, 1)
k_gi(const __bf16* __restrict__ xl,   const __bf16* __restrict__ xr,
     const __bf16* __restrict__ wihl, const __bf16* __restrict__ wihr,
     const float* __restrict__ bsl,   const float* __restrict__ bsr,
     float* __restrict__ gil, float* __restrict__ gir) {
  const int wave = blockIdx.x * (blockDim.x >> 5) + (threadIdx.x >> 5);
  const int dir = wave >> 14;          // 16384 wave-tiles per direction
  const int rem = wave & 16383;
  const int m0 = (rem >> 5) << 5;      // 512 M-supertiles of 32
  const int n0 = (rem & 31) << 6;      // 32 N-supertiles of 64
  const __bf16* X  = dir ? xr   : xl;
  const __bf16* W  = dir ? wihr : wihl;
  const float*  bs = dir ? bsr  : bsl;
  float*        G  = dir ? gir  : gil;

  v8f acc[2][4];
#pragma unroll
  for (int mi = 0; mi < 2; ++mi)
#pragma unroll
    for (int ni = 0; ni < 4; ++ni)
      acc[mi][ni] = v8f{0.f, 0.f, 0.f, 0.f, 0.f, 0.f, 0.f, 0.f};

  for (int k0 = 0; k0 < E_; k0 += 32) {
    bf16x16 a0 = load_tileK(X, E_, m0, k0);
    bf16x16 a1 = load_tileK(X, E_, m0 + 16, k0);
    bf16x16 b0 = load_tileK(W, E_, n0 +  0, k0);
    bf16x16 b1 = load_tileK(W, E_, n0 + 16, k0);
    bf16x16 b2 = load_tileK(W, E_, n0 + 32, k0);
    bf16x16 b3 = load_tileK(W, E_, n0 + 48, k0);
    acc[0][0] = wmma_bf16(a0, b0, acc[0][0]);
    acc[0][1] = wmma_bf16(a0, b1, acc[0][1]);
    acc[0][2] = wmma_bf16(a0, b2, acc[0][2]);
    acc[0][3] = wmma_bf16(a0, b3, acc[0][3]);
    acc[1][0] = wmma_bf16(a1, b0, acc[1][0]);
    acc[1][1] = wmma_bf16(a1, b1, acc[1][1]);
    acc[1][2] = wmma_bf16(a1, b2, acc[1][2]);
    acc[1][3] = wmma_bf16(a1, b3, acc[1][3]);
  }

  const int lane = threadIdx.x & 31;
  const int half = lane >> 4;
#pragma unroll
  for (int ni = 0; ni < 4; ++ni) {
    const int col = n0 + ni * 16 + (lane & 15);
    const float bv = bs[col];
#pragma unroll
    for (int mi = 0; mi < 2; ++mi) {
#pragma unroll
      for (int r = 0; r < 8; ++r) {
        int row = m0 + mi * 16 + r + (half << 3);
        G[(size_t)row * G4 + col] = acc[mi][ni][r] + bv;
      }
    }
  }
}

// ---- One LSTM time-step (both directions) ----------------------------------
// Block = (dir, 32 hidden cols): stages its Whh slice (2 n-tiles x 4 gates x
// 16 rows x K=512 bf16 = 128 KB) into dynamic LDS with
// global_load_async_to_lds_b128 (ASYNCcnt), then 8 waves (2 n-tiles x 4
// m-tiles) run the GEMM reading B from LDS (ds_load) and A (h_prev) from
// global. Each wave computes all four gate tiles for its 16x16 block so the
// gate nonlinearity is local. h double-buffered in bf16 across launches.

__global__ void __launch_bounds__(256, 1)
k_step(int s,
       const float* __restrict__ gil, const float* __restrict__ gir,
       const __bf16* __restrict__ whhl, const __bf16* __restrict__ whhr,
       const __bf16* __restrict__ h_in, __bf16* __restrict__ h_out,
       float* __restrict__ c, float* __restrict__ out) {
  extern __shared__ __bf16 Bs[];     // [128 rows][512 K] = 128 KB
  const int dir   = blockIdx.x >> 4;       // 2 dirs x 16 col-pairs = 32 blocks
  const int npair = blockIdx.x & 15;       // 32-column slice of H
  const __bf16* W = dir ? whhr : whhl;
  const float*  G = dir ? gir  : gil;

  // --- async-stage Whh slice: 8192 x 16B chunks, 32 per thread ---
  for (int ch = threadIdx.x; ch < 8192; ch += 256) {
    const int kc   = ch & 63;              // 16B chunk within a row (512 bf16)
    const int row  = ch >> 6;              // 0..127 LDS row
    const int nt   = row >> 6;             // 0..1   n-tile
    const int gate = (row >> 4) & 3;       // 0..3
    const int r    = row & 15;
    const __bf16* gp = W +
        ((size_t)(gate * H_ + npair * 32 + nt * 16 + r) * H_ + kc * 8);
    unsigned lds = (unsigned)(size_t)(&Bs[(size_t)row * H_ + kc * 8]);
    asm volatile("global_load_async_to_lds_b128 %0, %1, off"
                 :: "v"(lds), "v"(gp) : "memory");
  }
  asm volatile("s_wait_asynccnt 0" ::: "memory");
  __syncthreads();

  const int wv = threadIdx.x >> 5;         // 0..7
  const int ntl = wv >> 2;                 // 0..1 local n-tile
  const int mt  = wv & 3;                  // 0..3 batch tile
  const int m0  = mt << 4;
  const int n0  = npair * 32 + ntl * 16;   // global hidden col base
  const int lane = threadIdx.x & 31;
  const int half = lane >> 4;

  // B tile from LDS, same K-major striping as load_tileK.
  auto load_B_lds = [&](int rowbase, int k0) {
    const int rr = rowbase + (lane & 15);
    const int kk = k0 + (half << 3);
    bf16x16 v;
#pragma unroll
    for (int e = 0; e < 8; ++e) v[e] = Bs[(size_t)rr * H_ + kk + e];
#pragma unroll
    for (int e = 0; e < 8; ++e) v[8 + e] = Bs[(size_t)rr * H_ + kk + 16 + e];
    return v;
  };

  const __bf16* Hp = h_in + (size_t)dir * (B_ * H_);
  v8f ai = {0.f,0.f,0.f,0.f,0.f,0.f,0.f,0.f};
  v8f af = ai, ag = ai, ao = ai;
  for (int k0 = 0; k0 < H_; k0 += 32) {
    bf16x16 a  = load_tileK(Hp, H_, m0, k0);   // shared A (h_prev)
    bf16x16 b0 = load_B_lds(ntl * 64 +  0, k0);
    bf16x16 b1 = load_B_lds(ntl * 64 + 16, k0);
    bf16x16 b2 = load_B_lds(ntl * 64 + 32, k0);
    bf16x16 b3 = load_B_lds(ntl * 64 + 48, k0);
    ai = wmma_bf16(a, b0, ai);
    af = wmma_bf16(a, b1, af);
    ag = wmma_bf16(a, b2, ag);
    ao = wmma_bf16(a, b3, ao);
  }

  const int col = n0 + (lane & 15);
  float*  cp  = c     + (size_t)dir * (B_ * H_);
  __bf16* hop = h_out + (size_t)dir * (B_ * H_);
#pragma unroll
  for (int r = 0; r < 8; ++r) {
    int b = m0 + r + (half << 3);
    const float* grow = G + ((size_t)s * B_ + b) * G4;
    float iv = sigmoidf_(grow[         col] + ai[r]);
    float fv = sigmoidf_(grow[H_     + col] + af[r]);
    float gv = tanhf    (grow[2 * H_ + col] + ag[r]);
    float ov = sigmoidf_(grow[3 * H_ + col] + ao[r]);
    float cprev = cp[(size_t)b * H_ + col];
    float c2 = fv * cprev + iv * gv;
    float h2 = ov * tanhf(c2);
    cp [(size_t)b * H_ + col] = c2;
    hop[(size_t)b * H_ + col] = (__bf16)h2;
    out[((size_t)s * B_ + b) * (2 * H_) + (size_t)dir * H_ + col] = h2;
  }
}

// ---------------------------------------------------------------------------

extern "C" void kernel_launch(void* const* d_in, const int* in_sizes, int n_in,
                              void* d_out, int out_size, void* d_ws, size_t ws_size,
                              hipStream_t stream) {
  const int*   ic   = (const int*)  d_in[0];
  const int*   ibl  = (const int*)  d_in[1];
  // d_in[2] (insts_bichar_r) intentionally unused — reference uses _l twice.
  const float* cts  = (const float*)d_in[3];
  const float* ct   = (const float*)d_in[4];
  const float* bts  = (const float*)d_in[5];
  const float* bt   = (const float*)d_in[6];
  const float* Wl   = (const float*)d_in[7];
  const float* bl   = (const float*)d_in[8];
  const float* Wr   = (const float*)d_in[9];
  const float* br   = (const float*)d_in[10];
  const float* Wihl = (const float*)d_in[11];
  const float* Whhl = (const float*)d_in[12];
  const float* bihl = (const float*)d_in[13];
  const float* bhhl = (const float*)d_in[14];
  const float* Wihr = (const float*)d_in[15];
  const float* Whhr = (const float*)d_in[16];
  const float* bihr = (const float*)d_in[17];
  const float* bhhr = (const float*)d_in[18];
  float* out = (float*)d_out;

  // --- carve workspace ---
  char* ws = (char*)d_ws;
  size_t off = 0;
  auto carve = [&](size_t bytes) -> void* {
    void* p = ws + off;
    off = (off + bytes + 255) & ~(size_t)255;
    return p;
  };
  const int MROWS = S_ * B_;                       // 16384
  __bf16* xin    = (__bf16*)carve((size_t)MROWS * K_IN * 2);
  __bf16* wlb    = (__bf16*)carve((size_t)E_ * K_IN * 2);
  __bf16* wrb    = (__bf16*)carve((size_t)E_ * K_IN * 2);
  __bf16* wihlb  = (__bf16*)carve((size_t)G4 * E_ * 2);
  __bf16* wihrb  = (__bf16*)carve((size_t)G4 * E_ * 2);
  __bf16* whhlb  = (__bf16*)carve((size_t)G4 * H_ * 2);
  __bf16* whhrb  = (__bf16*)carve((size_t)G4 * H_ * 2);
  float*  bsl    = (float*) carve((size_t)G4 * 4);
  float*  bsr    = (float*) carve((size_t)G4 * 4);
  __bf16* xl     = (__bf16*)carve((size_t)MROWS * E_ * 2);
  __bf16* xr     = (__bf16*)carve((size_t)MROWS * E_ * 2);
  float*  gil    = (float*) carve((size_t)MROWS * G4 * 4);
  float*  gir    = (float*) carve((size_t)MROWS * G4 * 4);
  __bf16* hb0    = (__bf16*)carve((size_t)2 * B_ * H_ * 2);
  __bf16* hb1    = (__bf16*)carve((size_t)2 * B_ * H_ * 2);
  float*  cbuf   = (float*) carve((size_t)2 * B_ * H_ * 4);

  const int T = 256;
  auto blks = [](int n, int t) { return (n + t - 1) / t; };

  // --- weight prep ---
  k_pad_cvt<<<blks(E_ * K_IN, T), T, 0, stream>>>(Wl, wlb, E_, 400, K_IN);
  k_pad_cvt<<<blks(E_ * K_IN, T), T, 0, stream>>>(Wr, wrb, E_, 400, K_IN);
  k_cvt<<<blks(G4 * E_, T), T, 0, stream>>>(Wihl, wihlb, G4 * E_);
  k_cvt<<<blks(G4 * E_, T), T, 0, stream>>>(Wihr, wihrb, G4 * E_);
  k_cvt<<<blks(G4 * H_, T), T, 0, stream>>>(Whhl, whhlb, G4 * H_);
  k_cvt<<<blks(G4 * H_, T), T, 0, stream>>>(Whhr, whhrb, G4 * H_);
  k_bsum<<<blks(G4, T), T, 0, stream>>>(bihl, bhhl, bsl, G4);
  k_bsum<<<blks(G4, T), T, 0, stream>>>(bihr, bhhr, bsr, G4);
  k_zero_bf<<<blks(2 * B_ * H_, T), T, 0, stream>>>(hb0, 2 * B_ * H_);
  k_zero_bf<<<blks(2 * B_ * H_, T), T, 0, stream>>>(hb1, 2 * B_ * H_);
  k_zero_f<<<blks(2 * B_ * H_, T), T, 0, stream>>>(cbuf, 2 * B_ * H_);

  // --- embedding gather (time-major rows) ---
  k_embed<<<MROWS, 128, 0, stream>>>(ic, ibl, cts, ct, bts, bt, xin);

  // --- projection GEMM + tanh: 8192 waves (32x64 tiles) -> 1024 blocks ---
  k_proj<<<1024, 256, 0, stream>>>(xin, wlb, wrb, bl, br, xl, xr);

  // --- gate preactivations: 32768 waves (32x64 tiles) -> 4096 blocks ---
  k_gi<<<4096, 256, 0, stream>>>(xl, xr, wihlb, wihrb, bsl, bsr, gil, gir);

  // --- sequential LSTM scan: 32 blocks/step, 128 KB dynamic LDS each ---
  const size_t stepLds = (size_t)128 * H_ * sizeof(__bf16);  // 128 KB
  for (int s = 0; s < S_; ++s) {
    __bf16* hin  = (s & 1) ? hb1 : hb0;
    __bf16* hout = (s & 1) ? hb0 : hb1;
    k_step<<<32, 256, stepLds, stream>>>(s, gil, gir, whhlb, whhrb,
                                         hin, hout, cbuf, out);
  }
}
